// _RuntimeLongcatMoE_78752520339555
// MI455X (gfx1250) — matrix operand order repaired and checked by reading, not compiled, not run on hardware
//
#include <hip/hip_runtime.h>

#define T_ 4096
#define H_ 2048
#define E_ 32
#define EZ_ 48      // E + zero experts
#define I_ 1024
#define K_ 6
#define MT_ 32      // tokens per expert tile (2 x 16-row WMMA sub-tiles)
#define ROUTED_SCALE 2.5f

typedef __attribute__((ext_vector_type(16))) __bf16 v16bf;
typedef __attribute__((ext_vector_type(2)))  __bf16 v2bf;
typedef __attribute__((ext_vector_type(8)))  float  v8f;

union V16 {
  v16bf v;
  unsigned u[8];
  uint4 q[2];
};

// Hot-path pack: two fp32 -> bf16x2, 1 VALU per dword.
// If the toolchain has v_cvt_pk_bf16_f32 use it (RNE); otherwise a single
// v_perm_b32 packs the truncated high halves.
#if __has_builtin(__builtin_amdgcn_cvt_pk_bf16_f32)
static __device__ __forceinline__ unsigned pack2_bf16_fast(float lo, float hi) {
  v2bf r = __builtin_amdgcn_cvt_pk_bf16_f32(lo, hi);
  return __builtin_bit_cast(unsigned, r);
}
#else
static __device__ __forceinline__ unsigned pack2_bf16_fast(float lo, float hi) {
  unsigned a = __builtin_bit_cast(unsigned, lo);
  unsigned b = __builtin_bit_cast(unsigned, hi);
  // v_perm_b32: S1 supplies byte-sel 0..3, S0 supplies byte-sel 4..7
  return __builtin_amdgcn_perm(b, a, 0x07060302u);   // {hi16(b), hi16(a)}
}
#endif

// Staging-path pack (one-time per tile): rounded to nearest, ties away from zero.
static __device__ __forceinline__ unsigned pack2_bf16_rne(float lo, float hi) {
  unsigned a = __builtin_bit_cast(unsigned, lo) + 0x8000u;
  unsigned b = __builtin_bit_cast(unsigned, hi) + 0x8000u;
  return __builtin_amdgcn_perm(b, a, 0x07060302u);
}

static __device__ __forceinline__ unsigned short f2bf(float f) {
  unsigned u = __builtin_bit_cast(unsigned, f) + 0x8000u;
  return (unsigned short)(u >> 16);
}

// Convert 16 consecutive fp32 (16B-aligned) to a bf16 WMMA fragment.
static __device__ __forceinline__ v16bf cvt16f(const float* p) {
  V16 r;
  const float4* q = (const float4*)p;
#pragma unroll
  for (int j = 0; j < 4; ++j) {
    float4 f = q[j];
    r.u[2*j+0] = pack2_bf16_fast(f.x, f.y);
    r.u[2*j+1] = pack2_bf16_fast(f.z, f.w);
  }
  return r.v;
}

static __device__ __forceinline__ float fast_silu(float g) {
  return g * __builtin_amdgcn_rcpf(1.0f + __expf(-g));
}

// ---------------- kernel 1: reset per-expert counters ----------------
__global__ void zero_counts_kernel(int* __restrict__ counts) {
  if (threadIdx.x < E_) counts[threadIdx.x] = 0;
}

// ---------------- kernel 2: router + top-k dispatch ----------------
__global__ __launch_bounds__(64)
void router_kernel(const float* __restrict__ x, const float* __restrict__ rw,
                   const float* __restrict__ bias,
                   int* __restrict__ counts, int* __restrict__ tokList,
                   float* __restrict__ wtList, float* __restrict__ zerow) {
  __shared__ float xsr[H_];
  __shared__ float sc[EZ_];
  const int t = blockIdx.x;

  const float4* xrow = (const float4*)(x + (size_t)t * H_);
  for (int i = threadIdx.x; i < H_ / 4; i += 64) {
    float4 v = xrow[i];
    xsr[i*4+0] = v.x; xsr[i*4+1] = v.y; xsr[i*4+2] = v.z; xsr[i*4+3] = v.w;
  }
  __syncthreads();

  const int e = threadIdx.x;
  if (e < EZ_) {
    const float* w = rw + (size_t)e * H_;
    float acc = 0.0f;
    for (int h = 0; h < H_; h += 4)
      acc += xsr[h]*w[h] + xsr[h+1]*w[h+1] + xsr[h+2]*w[h+2] + xsr[h+3]*w[h+3];
    sc[e] = __builtin_amdgcn_rcpf(1.0f + __expf(-acc));   // sigmoid score
  }
  __syncthreads();

  if (threadIdx.x == 0) {
    unsigned long long used = 0ull;
    int ids[K_];
    float ssum = 0.0f;
    for (int k = 0; k < K_; ++k) {
      int bi = 0; float bv = -1e30f;
      for (int j = 0; j < EZ_; ++j) {
        if (used & (1ull << j)) continue;
        float v = sc[j] + bias[j];          // selection uses biased score
        if (v > bv) { bv = v; bi = j; }
      }
      used |= (1ull << bi);
      ids[k] = bi;
      ssum += sc[bi];                       // renorm uses unbiased score
    }
    const float scale = ROUTED_SCALE * __builtin_amdgcn_rcpf(ssum);
    float z = 0.0f;
    for (int k = 0; k < K_; ++k) {
      int id = ids[k];
      float w = sc[id] * scale;
      if (id >= E_) {
        z += w;                             // identity "zero expert"
      } else {
        int pos = atomicAdd(&counts[id], 1);
        tokList[(size_t)id * T_ + pos] = t;
        wtList[(size_t)id * T_ + pos] = w;
      }
    }
    zerow[t] = z;
  }
}

// ---------------- kernel 3: out = x * zero_weight (initializes out) ----------------
__global__ __launch_bounds__(256)
void zero_combine_kernel(const float* __restrict__ x, const float* __restrict__ zerow,
                         float* __restrict__ out) {
  size_t i = (size_t)blockIdx.x * 256 + threadIdx.x;   // over float4 elements
  if (i >= (size_t)T_ * (H_ / 4)) return;
  int t = (int)(i / (H_ / 4));
  float4 v = ((const float4*)x)[i];
  float z = zerow[t];
  float4 o; o.x = v.x*z; o.y = v.y*z; o.z = v.z*z; o.w = v.w*z;
  ((float4*)out)[i] = o;
}

// ---------------- kernel 4: grouped expert FFN (bf16 WMMA, 32-token tiles) ----------------
__global__ __launch_bounds__(256)
void expert_ffn_kernel(const float* __restrict__ x,
                       const float* __restrict__ w1,
                       const float* __restrict__ w3,
                       const float* __restrict__ w2,
                       const int*   __restrict__ counts,
                       const int*   __restrict__ tokList,
                       const float* __restrict__ wtList,
                       float* __restrict__ out) {
  __shared__ unsigned short xs[MT_][H_];    // 128 KB: 32-token x tile, bf16
  __shared__ unsigned short acts[MT_][I_];  //  64 KB: silu(gate)*up, bf16
  __shared__ int   toks[MT_];
  __shared__ float wts[MT_];

  const int e    = blockIdx.y;
  const int cnt  = counts[e];
  const int base = blockIdx.x * MT_;
  if (base >= cnt) return;
  const int mvalid = min(MT_, cnt - base);

  const int tid = threadIdx.x;
  if (tid < MT_) {
    int src = (tid < mvalid) ? (base + tid) : base;   // pad with row 0, weight 0
    toks[tid] = tokList[(size_t)e * T_ + src];
    wts[tid]  = (tid < mvalid) ? wtList[(size_t)e * T_ + base + tid] : 0.0f;
  }
  __syncthreads();

  // stage x tile -> LDS bf16 (8-byte packed stores, round-to-nearest, ties away)
  for (int i = tid; i < MT_ * (H_ / 4); i += 256) {
    int m = i >> 9;                 // H_/4 == 512
    int c = (i & 511) * 4;
    float4 v = *(const float4*)(x + (size_t)toks[m] * H_ + c);
    uint2 p;
    p.x = pack2_bf16_rne(v.x, v.y);
    p.y = pack2_bf16_rne(v.z, v.w);
    *(uint2*)&xs[m][c] = p;
  }
  __syncthreads();

  const int lane = tid & 31;
  const int wv   = tid >> 5;        // wave id, 0..7
  const int half = lane >> 4;       // lane group
  const int mN   = lane & 15;       // A row / B column index
  const int koff = half * 16;       // 16-bit A-fragment k offset (contiguous k = 16h + s)

  // ---- Phase A: gate = x*w1^T, up = x*w3^T, act = silu(gate)*up ----
  const unsigned short* arow0 = &xs[mN][0];        // rows 0..15
  const unsigned short* arow1 = &xs[16 + mN][0];   // rows 16..31
#pragma unroll 1
  for (int nt = wv; nt < I_ / 16; nt += 8) {
    const int nbase = nt * 16;
    const float* w1row = w1 + ((size_t)e * I_ + nbase + mN) * H_;
    const float* w3row = w3 + ((size_t)e * I_ + nbase + mN) * H_;
    v8f g0 = {0,0,0,0,0,0,0,0}, g1 = {0,0,0,0,0,0,0,0};
    v8f u0 = {0,0,0,0,0,0,0,0}, u1 = {0,0,0,0,0,0,0,0};
    for (int k0 = 0; k0 < H_; k0 += 32) {
      V16 a0, a1;
      a0.q[0] = *(const uint4*)(arow0 + k0 + koff);
      a0.q[1] = *(const uint4*)(arow0 + k0 + koff + 8);
      a1.q[0] = *(const uint4*)(arow1 + k0 + koff);
      a1.q[1] = *(const uint4*)(arow1 + k0 + koff + 8);
      const int c0 = k0 + half * 16;
      __builtin_prefetch(w1row + c0 + 64, 0, 1);
      __builtin_prefetch(w3row + c0 + 64, 0, 1);
      v16bf b1 = cvt16f(w1row + c0);   // shared by both M sub-tiles
      v16bf b2 = cvt16f(w3row + c0);
      g0 = __builtin_amdgcn_wmma_f32_16x16x32_bf16(false, a0.v, false, b1, (short)0, g0, false, false);
      u0 = __builtin_amdgcn_wmma_f32_16x16x32_bf16(false, a0.v, false, b2, (short)0, u0, false, false);
      g1 = __builtin_amdgcn_wmma_f32_16x16x32_bf16(false, a1.v, false, b1, (short)0, g1, false, false);
      u1 = __builtin_amdgcn_wmma_f32_16x16x32_bf16(false, a1.v, false, b2, (short)0, u1, false, false);
    }
#pragma unroll
    for (int r = 0; r < 8; ++r) {
      int m = r + half * 8;                              // C-fragment row map
      acts[m][nbase + mN]      = f2bf(fast_silu(g0[r]) * u0[r]);
      acts[16 + m][nbase + mN] = f2bf(fast_silu(g1[r]) * u1[r]);
    }
  }
  __syncthreads();

  // ---- Phase B: y = act * w2^T, scale by routing weight, scatter-add ----
  const unsigned short* brow0 = &acts[mN][0];
  const unsigned short* brow1 = &acts[16 + mN][0];
#pragma unroll 1
  for (int nt = wv; nt < H_ / 16; nt += 8) {
    const int hbase = nt * 16;
    const float* w2row = w2 + ((size_t)e * H_ + hbase + mN) * I_;
    v8f acc0 = {0,0,0,0,0,0,0,0}, acc1 = {0,0,0,0,0,0,0,0};
    for (int k0 = 0; k0 < I_; k0 += 32) {
      V16 a0, a1;
      a0.q[0] = *(const uint4*)(brow0 + k0 + koff);
      a0.q[1] = *(const uint4*)(brow0 + k0 + koff + 8);
      a1.q[0] = *(const uint4*)(brow1 + k0 + koff);
      a1.q[1] = *(const uint4*)(brow1 + k0 + koff + 8);
      const int c0 = k0 + half * 16;
      __builtin_prefetch(w2row + c0 + 64, 0, 1);
      v16bf b = cvt16f(w2row + c0);
      acc0 = __builtin_amdgcn_wmma_f32_16x16x32_bf16(false, a0.v, false, b, (short)0, acc0, false, false);
      acc1 = __builtin_amdgcn_wmma_f32_16x16x32_bf16(false, a1.v, false, b, (short)0, acc1, false, false);
    }
#pragma unroll
    for (int r = 0; r < 8; ++r) {
      int m = r + half * 8;
      if (m < mvalid)
        atomicAdd(out + (size_t)toks[m] * H_ + hbase + mN, wts[m] * acc0[r]);
      if (16 + m < mvalid)
        atomicAdd(out + (size_t)toks[16 + m] * H_ + hbase + mN, wts[16 + m] * acc1[r]);
    }
  }
}

extern "C" void kernel_launch(void* const* d_in, const int* in_sizes, int n_in,
                              void* d_out, int out_size, void* d_ws, size_t ws_size,
                              hipStream_t stream) {
  const float* x  = (const float*)d_in[0];   // [T,H]
  const float* rw = (const float*)d_in[1];   // [E+Z,H]
  const float* cb = (const float*)d_in[2];   // [E+Z]
  const float* w1 = (const float*)d_in[3];   // [E,I,H]
  const float* w3 = (const float*)d_in[4];   // [E,I,H]
  const float* w2 = (const float*)d_in[5];   // [E,H,I]
  float* out = (float*)d_out;

  char* ws = (char*)d_ws;
  int*   counts  = (int*)ws;                                     // E ints
  int*   tokList = (int*)(ws + 256);                             // E*T ints
  float* wtList  = (float*)(ws + 256 + (size_t)E_ * T_ * 4);     // E*T floats
  float* zerow   = (float*)(ws + 256 + 2 * (size_t)E_ * T_ * 4); // T floats

  zero_counts_kernel<<<dim3(1), dim3(64), 0, stream>>>(counts);
  router_kernel<<<dim3(T_), dim3(64), 0, stream>>>(x, rw, cb, counts, tokList, wtList, zerow);
  zero_combine_kernel<<<dim3((T_ * (H_ / 4) + 255) / 256), dim3(256), 0, stream>>>(x, zerow, out);
  expert_ffn_kernel<<<dim3(T_ / MT_, E_), dim3(256), 0, stream>>>(x, w1, w3, w2, counts, tokList, wtList, out);
}
